// TileWarping1_87608742904186
// MI455X (gfx1250) — compile-verified, matrix operand order checked by reading, not run on hardware
//
#include <hip/hip_runtime.h>

// Problem constants (fixed by the reference setup_inputs).
#define B_   4
#define C_   32
#define H_   512
#define W_   960
#define TS_  4
#define HT_  (H_/TS_)            // 128
#define WT_  (W_/TS_)            // 240
#define PIX_PER_WAVE 16
#define WAVES_X (W_/PIX_PER_WAVE) // 60

typedef __attribute__((ext_vector_type(16))) _Float16 v16h;
typedef __attribute__((ext_vector_type(8)))  float    v8f;

__device__ __forceinline__ int clampx(int v) {
    return v < 0 ? 0 : (v > (W_ - 1) ? (W_ - 1) : v);
}

// One wave32 handles 16 consecutive x-pixels at one y for one batch image:
//   lanes 0-15  -> pixel m = lane,      channels 0..15
//   lanes 16-31 -> pixel m = lane-16,   channels 16..31
// (contiguous k = 16h + s, not AMD's table order; any k order works with all-ones B).
// Three A matrices (one per disparity offset) are reduced over K=32 channels
// by multiplying with an all-ones f16 B matrix; a second WMMA with the f16
// residual term recovers near-f32 accuracy.
__global__ __launch_bounds__(256) void tile_warp_cost_wmma(
    const float* __restrict__ tile_plane,   // [B,3,Ht,Wt]
    const float* __restrict__ fea_l,        // [B,C,H,W]
    const float* __restrict__ fea_r,        // [B,C,H,W]
    float* __restrict__ out)                // [B,48,Ht,Wt]
{
    const int lane        = threadIdx.x & 31;
    const int wave_in_blk = threadIdx.x >> 5;
    const int wave        = blockIdx.x * (blockDim.x >> 5) + wave_in_blk;
    const int total_waves = B_ * H_ * WAVES_X;
    if (wave >= total_waves) return;        // wave-uniform (never diverges)

    const int xw = wave % WAVES_X;
    const int y  = (wave / WAVES_X) % H_;
    const int b  = wave / (WAVES_X * H_);

    const int x_base = xw * PIX_PER_WAVE;
    const int pix    = lane & 15;           // matrix row m (pixel within group)
    const int cbase  = (lane >> 4) * 16;    // channel base for this lane
    const int x      = x_base + pix;
    const int ht     = y >> 2;
    const int iy     = y & 3;               // intra-tile row
    const int jx     = pix & 3;             // intra-tile col (x_base % 4 == 0)
    const int wt     = x >> 2;

    // ---- slanted-plane disparity for this pixel -------------------------
    const int   tb   = (b * 3 * HT_ + ht) * WT_ + wt;
    const float d    = tile_plane[tb];
    const float dx   = tile_plane[tb +     HT_ * WT_];
    const float dy   = tile_plane[tb + 2 * HT_ * WT_];
    const float disp0 = d + ((float)iy - 1.5f) * dy + ((float)jx - 1.5f) * dx;

    const float xs  = (float)x - disp0;     // source x for delta = 0
    const float x0f = floorf(xs);
    const float w   = xs - x0f;             // shared fractional weight
    const int   x0  = (int)x0f;

    // delta = -1 samples xs+1, delta = +1 samples xs-1: all three lerps use
    // the 4 consecutive taps fea_r[x0-1 .. x0+2].
    const int im1 = clampx(x0 - 1);
    const int i0c = clampx(x0);
    const int i1c = clampx(x0 + 1);
    const int i2c = clampx(x0 + 2);

    const float wlim = (float)(W_ - 1);
    const bool vA = (xs + 1.0f >= 0.0f) && (xs + 1.0f <= wlim);  // delta=-1
    const bool vB = (xs        >= 0.0f) && (xs        <= wlim);  // delta= 0
    const bool vC = (xs - 1.0f >= 0.0f) && (xs - 1.0f <= wlim);  // delta=+1

    const size_t HW   = (size_t)H_ * W_;
    const size_t base = ((size_t)(b * C_ + cbase) * H_ + y) * (size_t)W_;
    const float* fl_p = fea_l + base + x;
    const float* fr_p = fea_r + base;

    // ---- build A matrices (hi + residual f16 split) ---------------------
    v16h aAh, aAl, aBh, aBl, aCh, aCl;
#pragma unroll
    for (int cc = 0; cc < 16; ++cc) {
        const size_t co = (size_t)cc * HW;
        const float fl  = fl_p[co];
        const float fm1 = fr_p[co + im1];
        const float f0  = fr_p[co + i0c];
        const float f1  = fr_p[co + i1c];
        const float f2  = fr_p[co + i2c];

        float wA = f1  + w * (f2 - f1);     // delta=-1 (floor = x0+1)
        float wB = f0  + w * (f1 - f0);     // delta= 0 (floor = x0)
        float wC = fm1 + w * (f0 - fm1);    // delta=+1 (floor = x0-1)
        wA = vA ? wA : 0.0f;
        wB = vB ? wB : 0.0f;
        wC = vC ? wC : 0.0f;

        const float dAv = fabsf(fl - wA);
        const float dBv = fabsf(fl - wB);
        const float dCv = fabsf(fl - wC);

        const _Float16 hA = (_Float16)dAv;
        const _Float16 hB = (_Float16)dBv;
        const _Float16 hC = (_Float16)dCv;
        aAh[cc] = hA;  aAl[cc] = (_Float16)(dAv - (float)hA);
        aBh[cc] = hB;  aBl[cc] = (_Float16)(dBv - (float)hB);
        aCh[cc] = hC;  aCl[cc] = (_Float16)(dCv - (float)hC);
    }

    // All-ones B: D[m][n] = sum_k A[m][k] regardless of B's lane layout.
    v16h bones;
#pragma unroll
    for (int e = 0; e < 16; ++e) bones[e] = (_Float16)1.0f;

    v8f zero = {};
    // args: (neg_a, A, neg_b, B, c_mod, C, reuse_a, reuse_b)
    v8f dA = __builtin_amdgcn_wmma_f32_16x16x32_f16(false, aAh, false, bones, (short)0, zero, false, false);
    dA     = __builtin_amdgcn_wmma_f32_16x16x32_f16(false, aAl, false, bones, (short)0, dA,   false, false);
    v8f dB = __builtin_amdgcn_wmma_f32_16x16x32_f16(false, aBh, false, bones, (short)0, zero, false, false);
    dB     = __builtin_amdgcn_wmma_f32_16x16x32_f16(false, aBl, false, bones, (short)0, dB,   false, false);
    v8f dC = __builtin_amdgcn_wmma_f32_16x16x32_f16(false, aCh, false, bones, (short)0, zero, false, false);
    dC     = __builtin_amdgcn_wmma_f32_16x16x32_f16(false, aCl, false, bones, (short)0, dC,   false, false);

    // ---- scatter to tile-unshuffled output ------------------------------
    // D layout: VGPR r holds M=r (lanes 0-15) / M=r+8 (lanes 16-31).
    // Lanes 0-7 store pixels 0-7 (slot = lane), lanes 16-23 store pixels
    // 8-15 (slot = lane-16).  out[b][16*di + iy*4 + p%4][ht][wt0 + p/4].
    const int  sel    = lane & 7;
    const bool active = (lane & 15) < 8;
    const int  p      = (lane < 16) ? lane : (lane - 8);

    float sA = dA[0], sB = dB[0], sC = dC[0];
#pragma unroll
    for (int k = 1; k < 8; ++k) {
        if (sel == k) { sA = dA[k]; sB = dB[k]; sC = dC[k]; }
    }

    if (active) {
        const int pj  = p & 3;
        const int pwt = (x_base >> 2) + (p >> 2);
        const size_t plane = (size_t)HT_ * WT_;
        const size_t o0 = (((size_t)b * 48 + (size_t)(iy * 4 + pj)) * HT_ + ht) * WT_ + pwt;
        out[o0]              = sA;   // delta = -1 -> channels  0..15
        out[o0 + 16 * plane] = sB;   // delta =  0 -> channels 16..31
        out[o0 + 32 * plane] = sC;   // delta = +1 -> channels 32..47
    }
}

extern "C" void kernel_launch(void* const* d_in, const int* in_sizes, int n_in,
                              void* d_out, int out_size, void* d_ws, size_t ws_size,
                              hipStream_t stream) {
    (void)in_sizes; (void)n_in; (void)out_size; (void)d_ws; (void)ws_size;
    const float* tile_plane = (const float*)d_in[0];
    const float* fea_l      = (const float*)d_in[1];
    const float* fea_r      = (const float*)d_in[2];
    float*       out        = (float*)d_out;

    const int total_waves    = B_ * H_ * WAVES_X;   // 122880
    const int waves_per_blk  = 8;                   // 256 threads, wave32
    const int blocks         = (total_waves + waves_per_blk - 1) / waves_per_blk;

    hipLaunchKernelGGL(tile_warp_cost_wmma, dim3(blocks), dim3(32 * waves_per_blk),
                       0, stream, tile_plane, fea_l, fea_r, out);
}
